// MultiHeadedAttentionLayer_9070970929509
// MI455X (gfx1250) — compile-verified, matrix operand order checked by reading, not run on hardware
//
#include <hip/hip_runtime.h>
#include <hip/hip_bf16.h>

typedef __attribute__((ext_vector_type(16))) __bf16 v16bf;
typedef __attribute__((ext_vector_type(2)))  __bf16 v2bf;
typedef __attribute__((ext_vector_type(8)))  float  v8f;

union BfOp { v16bf v; unsigned u[8]; };

// ---- bf16 conversion: let ISel pick the HW convert (v_cvt_pk_bf16_f32 /
// v_cvt_bf16_f32) via fptrunc; falls back to inline RNE expansion otherwise.
__device__ __forceinline__ unsigned pack2(float lo, float hi) {
#if __has_builtin(__builtin_amdgcn_cvt_pk_bf16_f32)
  return __builtin_bit_cast(unsigned, __builtin_amdgcn_cvt_pk_bf16_f32(lo, hi));
#else
  v2bf t = { (__bf16)lo, (__bf16)hi };
  static_assert(sizeof(v2bf) == 4, "v2bf must be 32-bit");
  return __builtin_bit_cast(unsigned, t);
#endif
}
__device__ __forceinline__ unsigned short f2bf(float f) {
  return __builtin_bit_cast(unsigned short, (__bf16)f);
}

constexpr int Bb = 2, Hh = 16, Mm = 4096, Dd = 64, NB = 64;   // WM = WN = 64
constexpr int KSTR = 66;    // K LDS row stride (halfwords), odd dword stride
constexpr int PSTR = 198;   // P / Vt LDS row stride (halfwords)

// Core for one q-block strip. NT = #16-wide key tiles (8 or 12). Fully unrolled
// so accumulators stay in VGPRs. Ks/Ps alias (barrier after the S GEMM).
template<int NT, bool MIDDLE>
__device__ __forceinline__ void attn_core(
    const unsigned short* __restrict__ Ks,   // [16*NT][KSTR] bf16, rows = key, cols = d
    unsigned short*       __restrict__ Ps,   // [64][PSTR]    bf16, rows = m,   cols = key (aliases Ks)
    const unsigned short* __restrict__ Vt,   // [64][PSTR]    bf16, rows = d,   cols = key
    const float* __restrict__ qg,            // Q block base (64 x 64, row-major)
    const float* __restrict__ bmp,           // band_mask slice [64][192] (MIDDLE only)
    const float* __restrict__ tmsk,          // to_mask + b*M + gcol0     (!MIDDLE)
    const float* __restrict__ fmp,           // from_mask + b*M + l*64
    float*       __restrict__ outp)          // out + ((b*M + l*64)*H + h)*D
{
  constexpr int NC = NT / 2;                 // #32-deep k-chunks for O GEMM
  const int tid  = threadIdx.x;
  const int lane = tid & 31, wave = tid >> 5;
  const int hf   = lane >> 4, l16 = lane & 15;

  int dmap[8];
  #pragma unroll
  for (int j = 0; j < 8; ++j)
    dmap[j] = (j < 4) ? (8 * hf + 2 * j) : (16 + 8 * hf + 2 * (j - 4));

  // ---- A-operands: this wave's 16 Q rows, scaled, bf16, direct from global ----
  const float scale = 0.125f;                // 1/sqrt(64)
  BfOp qa[2];
  const int qrow = 16 * wave + l16;
  #pragma unroll
  for (int c = 0; c < 2; ++c)
    #pragma unroll
    for (int j = 0; j < 8; ++j) {
      const int d = 32 * c + dmap[j];
      const float x = qg[qrow * Dd + d], y = qg[qrow * Dd + d + 1];
      qa[c].u[j] = pack2(x * scale, y * scale);
    }

  // ---- S = (Q*scale) . K^T : NT tiles of 16x16, contraction d = 2 x k32 ----
  v8f sacc[NT];
  #pragma unroll
  for (int t = 0; t < NT; ++t) {
    v8f acc = {};
    const int krow = t * 16 + l16;
    #pragma unroll
    for (int c = 0; c < 2; ++c) {
      BfOp bop;
      #pragma unroll
      for (int j = 0; j < 8; ++j)
        bop.u[j] = *(const unsigned*)(Ks + krow * KSTR + 32 * c + dmap[j]);
      acc = __builtin_amdgcn_wmma_f32_16x16x32_bf16(false, qa[c].v, false, bop.v,
                                                    (short)0, acc, false, false);
    }
    sacc[t] = acc;
  }
  __syncthreads();   // every wave done reading Ks; region may be reused as Ps

  // ---- additive mask + row-wise softmax (C-layout: row m=r+8*hf on VGPR r) ----
  float rmax[8], rsum[8];
  #pragma unroll
  for (int r = 0; r < 8; ++r) rmax[r] = -3.0e38f;
  #pragma unroll
  for (int t = 0; t < NT; ++t) {
    const int col = t * 16 + l16;
    if constexpr (MIDDLE) {
      #pragma unroll
      for (int r = 0; r < 8; ++r) {
        const int mrow = 16 * wave + r + 8 * hf;
        sacc[t][r] += (1.0f - bmp[(size_t)mrow * 192 + col]) * -10000.0f;
      }
    } else {
      const float a = (1.0f - tmsk[col]) * -10000.0f;
      #pragma unroll
      for (int r = 0; r < 8; ++r) sacc[t][r] += a;
    }
    #pragma unroll
    for (int r = 0; r < 8; ++r) rmax[r] = fmaxf(rmax[r], sacc[t][r]);
  }
  #pragma unroll
  for (int m = 1; m <= 8; m <<= 1)
    #pragma unroll
    for (int r = 0; r < 8; ++r)
      rmax[r] = fmaxf(rmax[r], __shfl_xor(rmax[r], m, 32));   // within 16-lane half
  #pragma unroll
  for (int r = 0; r < 8; ++r) rsum[r] = 0.0f;
  #pragma unroll
  for (int t = 0; t < NT; ++t)
    #pragma unroll
    for (int r = 0; r < 8; ++r) {
      const float p = __expf(sacc[t][r] - rmax[r]);
      sacc[t][r] = p;
      rsum[r] += p;
    }
  #pragma unroll
  for (int m = 1; m <= 8; m <<= 1)
    #pragma unroll
    for (int r = 0; r < 8; ++r)
      rsum[r] += __shfl_xor(rsum[r], m, 32);

  // ---- stage P (bf16) in LDS; strip is private to this wave ----
  #pragma unroll
  for (int t = 0; t < NT; ++t) {
    const int col = t * 16 + l16;
    #pragma unroll
    for (int r = 0; r < 8; ++r)
      Ps[(16 * wave + r + 8 * hf) * PSTR + col] = f2bf(sacc[t][r]);
  }
  asm volatile("s_wait_dscnt 0" ::: "memory");  // in-wave store->load ordering

  // ---- O = P . V (contraction over key index, NC chunks of 32) ----
  v8f oacc[4] = {{}, {}, {}, {}};
  #pragma unroll
  for (int c = 0; c < NC; ++c) {
    BfOp pa;
    const int prow = 16 * wave + l16;
    #pragma unroll
    for (int j = 0; j < 8; ++j)
      pa.u[j] = *(const unsigned*)(Ps + prow * PSTR + 32 * c + dmap[j]);
    #pragma unroll
    for (int dt = 0; dt < 4; ++dt) {
      BfOp vb;
      const int dcol = dt * 16 + l16;
      #pragma unroll
      for (int j = 0; j < 8; ++j)
        vb.u[j] = *(const unsigned*)(Vt + dcol * PSTR + 32 * c + dmap[j]);
      oacc[dt] = __builtin_amdgcn_wmma_f32_16x16x32_bf16(false, pa.v, false, vb.v,
                                                         (short)0, oacc[dt], false, false);
    }
  }

  // ---- epilogue: 1/rowsum, from_mask, scatter to (B, M, H, D) ----
  #pragma unroll
  for (int r = 0; r < 8; ++r) rsum[r] = 1.0f / rsum[r];
  #pragma unroll
  for (int dt = 0; dt < 4; ++dt)
    #pragma unroll
    for (int r = 0; r < 8; ++r) {
      const int mrow = 16 * wave + r + 8 * hf;
      outp[(size_t)mrow * Hh * Dd + dt * 16 + l16] =
          oacc[dt][r] * rsum[r] * fmp[mrow];
    }
}

__global__ __launch_bounds__(128)
void bigbird_attn_kernel(const float* __restrict__ Q, const float* __restrict__ K,
                         const float* __restrict__ V, const float* __restrict__ band_mask,
                         const float* __restrict__ from_mask, const float* __restrict__ to_mask,
                         float* __restrict__ out) {
  __shared__ unsigned short smem[192 * KSTR + 64 * PSTR];   // ~49.5 KB
  unsigned short* Ks = smem;                 // [<=192][KSTR]  (later reused as Ps)
  unsigned short* Ps = smem;                 // [64][PSTR]
  unsigned short* Vt = smem + 192 * KSTR;    // [64][PSTR]

  const int wg = blockIdx.x;
  const int l  = wg % NB;
  const int bh = wg / NB;
  const int b  = bh / Hh, h = bh % Hh;

  int Kc, k0; const float* vsrc;
  if (l == 0)           { Kc = 128; k0 = 0;            vsrc = K; }  // first_v = bk (ref quirk)
  else if (l == NB - 1) { Kc = 128; k0 = Mm - 128;     vsrc = K; }  // last_v = bk (ref quirk)
  else                  { Kc = 192; k0 = (l - 1) * 64; vsrc = V; }

  const size_t base = (size_t)(b * Hh + h) * Mm * Dd;
  const float4* kg = (const float4*)(K    + base + (size_t)k0 * Dd);
  const float4* vg = (const float4*)(vsrc + base + (size_t)k0 * Dd);
  const float*  qg = Q + base + (size_t)l * 64 * Dd;

  // Cooperative staging by ROW PAIRS so every bf16 convert is packed and every
  // transposed-V store is a 4-byte ds_store (adjacent keys share a dword).
  for (int p = threadIdx.x; p < Kc * 8; p += 128) {
    const int rp = p >> 4;                   // key-row pair index
    const int c4 = (p & 15) << 2;            // starting d (multiple of 4)
    const int r0 = rp * 2, r1 = r0 + 1;
    const float4 k0 = kg[(size_t)r0 * 16 + (c4 >> 2)];
    const float4 k1 = kg[(size_t)r1 * 16 + (c4 >> 2)];
    *(unsigned*)(Ks + r0 * KSTR + c4)     = pack2(k0.x, k0.y);
    *(unsigned*)(Ks + r0 * KSTR + c4 + 2) = pack2(k0.z, k0.w);
    *(unsigned*)(Ks + r1 * KSTR + c4)     = pack2(k1.x, k1.y);
    *(unsigned*)(Ks + r1 * KSTR + c4 + 2) = pack2(k1.z, k1.w);
    const float4 v0 = vg[(size_t)r0 * 16 + (c4 >> 2)];
    const float4 v1 = vg[(size_t)r1 * 16 + (c4 >> 2)];
    *(unsigned*)(Vt + (c4 + 0) * PSTR + r0) = pack2(v0.x, v1.x);
    *(unsigned*)(Vt + (c4 + 1) * PSTR + r0) = pack2(v0.y, v1.y);
    *(unsigned*)(Vt + (c4 + 2) * PSTR + r0) = pack2(v0.z, v1.z);
    *(unsigned*)(Vt + (c4 + 3) * PSTR + r0) = pack2(v0.w, v1.w);
  }
  __syncthreads();

  const float* fmp  = from_mask + (size_t)b * Mm + (size_t)l * 64;
  float*       outp = out + (((size_t)b * Mm + (size_t)l * 64) * Hh + h) * Dd;

  if (l > 0 && l < NB - 1) {
    const float* bmp = band_mask + (size_t)(b * (NB - 2) + (l - 1)) * 64 * 192;
    attn_core<12, true >(Ks, Ps, Vt, qg, bmp, nullptr, fmp, outp);
  } else {
    const int gcol0 = (l == 0) ? 0 : (Mm - 128);
    const float* tmsk = to_mask + (size_t)b * Mm + gcol0;
    attn_core< 8, false>(Ks, Ps, Vt, qg, nullptr, tmsk, fmp, outp);
  }
}

extern "C" void kernel_launch(void* const* d_in, const int* in_sizes, int n_in,
                              void* d_out, int out_size, void* d_ws, size_t ws_size,
                              hipStream_t stream) {
  (void)in_sizes; (void)n_in; (void)out_size; (void)d_ws; (void)ws_size;
  const float* Q  = (const float*)d_in[0];
  const float* K  = (const float*)d_in[1];
  const float* V  = (const float*)d_in[2];
  const float* bm = (const float*)d_in[3];
  const float* fm = (const float*)d_in[4];
  const float* tm = (const float*)d_in[5];
  float* out = (float*)d_out;
  dim3 grid(Bb * Hh * NB);   // 2048 workgroups, one per (b, h, q-block)
  dim3 block(128);           // 4 waves; each wave owns a 16-row Q strip
  bigbird_attn_kernel<<<grid, block, 0, stream>>>(Q, K, V, bm, fm, tm, out);
}
